// ExecutableModel_34342558498846
// MI455X (gfx1250) — matrix operand order repaired and checked by reading, not compiled, not run on hardware
//
#include <hip/hip_runtime.h>
#include <math.h>

#define TOKENS  2048
#define S_LEN   1024
#define DMODEL  1024
#define NHEADS  16
#define DHEAD   64
#define FFDIM   4096
#define VOCAB   32000
#define NLAYERS 4

typedef __bf16 bf16;
typedef __attribute__((ext_vector_type(16))) __bf16 v16bf;
typedef __attribute__((ext_vector_type(8)))  float  v8f;
typedef int si4 __attribute__((ext_vector_type(4)));
typedef int si8 __attribute__((ext_vector_type(8)));

// ---------------------------------------------------------------------------
// WMMA fragment loaders (CDNA5 wave32 layouts, CDNA5 ISA §7.12.2)
// ---------------------------------------------------------------------------
__device__ __forceinline__ v16bf load_a_frag(const bf16* tile, int rowStride, int lane) {
    int m   = lane & 15;
    int kp0 = (lane >> 4) << 3;               // +8 u32 pairs (= +16 k) upper half
    const unsigned* p = (const unsigned*)(tile + (size_t)m * rowStride);
    union { v16bf v; unsigned u[8]; } r;
#pragma unroll
    for (int q = 0; q < 8; ++q)
        r.u[q] = p[q + kp0];
    return r.v;
}

__device__ __forceinline__ v16bf load_b_frag(const bf16* tile, int rowStride, int lane) {
    int n  = lane & 15;
    int k0 = (lane >> 4) << 4;                // +16 k for upper half lanes
    const unsigned* p = (const unsigned*)(tile + (size_t)n * rowStride + k0);
    union { v16bf v; unsigned u[8]; } r;
#pragma unroll
    for (int q = 0; q < 8; ++q) r.u[q] = p[q];
    return r.v;
}

// ---------------------------------------------------------------------------
// Tensor Data Mover: async 2D tile (rows x 32 bf16 elements) global -> LDS.
// D# layout per CDNA5 ISA. Groups passed in SGPR
// tuples; groups 2/3 zero (2D tile, tile_dim2/3 = 0). Tracked by TENSORcnt.
// ---------------------------------------------------------------------------
__device__ __forceinline__ void tdm_load_tile_2d(unsigned lds_byte_off,
                                                 const bf16* gptr,
                                                 unsigned rows,
                                                 unsigned row_stride_elems)
{
    unsigned long long ga = (unsigned long long)(size_t)gptr;
    si4 g0;
    g0[0] = 1;                                             // count=1 (valid user D#)
    g0[1] = (int)lds_byte_off;                             // lds_addr
    g0[2] = (int)(unsigned)(ga & 0xffffffffull);           // global_addr[31:0]
    g0[3] = (int)((unsigned)((ga >> 32) & 0x01ffffffull)   // global_addr[56:32]
                  | 0x80000000u);                          // type=2 ("image")
    si8 g1;
    unsigned d0 = 32u;                                     // tensor_dim0 (= tile_dim0)
    unsigned d1 = rows;                                    // tensor_dim1 (= tile_dim1)
    g1[0] = (int)(1u << 16);                               // data_size=1 (2 bytes)
    g1[1] = (int)((d0 & 0xffffu) << 16);                   // tensor_dim0[15:0]
    g1[2] = (int)((d0 >> 16) | ((d1 & 0xffffu) << 16));    // dim0 hi / dim1 lo
    g1[3] = (int)((d1 >> 16) | (32u << 16));               // dim1 hi / tile_dim0=32
    g1[4] = (int)(rows & 0xffffu);                         // tile_dim1; tile_dim2=0
    g1[5] = (int)row_stride_elems;                         // tensor_dim0_stride lo32
    g1[6] = 0;                                             // stride0 hi / stride1 lo
    g1[7] = 0;
    si4 gz = {0, 0, 0, 0};
    asm volatile("tensor_load_to_lds %0, %1, %2, %3"
                 :: "s"(g0), "s"(g1), "s"(gz), "s"(gz)
                 : "memory");
}

// ---------------------------------------------------------------------------
// Generic bf16 GEMM: C[M,N] = A[M,K] @ W[N,K]^T (+bias)(+gelu)(+res)
// 256 threads = 8 waves; 128x128 block tile; BK=32; wave tile 32x64.
// Tiles streamed by TDM, double-buffered: issue k+1, wait tile k, compute.
// ---------------------------------------------------------------------------
__global__ __launch_bounds__(256)
void gemm_bf16_kernel(const bf16* __restrict__ A, const bf16* __restrict__ W,
                      const float* __restrict__ bias, const float* __restrict__ res,
                      float* __restrict__ outF, bf16* __restrict__ outB,
                      int M, int N, int K, int act)
{
    __shared__ __align__(16) unsigned As[2][128 * 16];
    __shared__ __align__(16) unsigned Ws[2][128 * 16];

    int tid = threadIdx.x;
    int lane = tid & 31, wave = tid >> 5;
    int waveM = wave >> 1, waveN = wave & 1;
    int bm = blockIdx.y * 128, bn = blockIdx.x * 128;

    v8f vzero = {0.f, 0.f, 0.f, 0.f, 0.f, 0.f, 0.f, 0.f};
    v8f acc[2][4];
#pragma unroll
    for (int i = 0; i < 2; ++i)
#pragma unroll
        for (int j = 0; j < 4; ++j) acc[i][j] = vzero;

    int nk = K >> 5;
    if (wave == 0) {
        tdm_load_tile_2d((unsigned)(size_t)&As[0][0], A + (size_t)bm * K, 128, K);
        tdm_load_tile_2d((unsigned)(size_t)&Ws[0][0], W + (size_t)bn * K, 128, K);
    }

    for (int it = 0; it < nk; ++it) {
        int buf = it & 1;
        if (wave == 0) {
            if (it + 1 < nk) {
                int k0n = (it + 1) << 5;
                tdm_load_tile_2d((unsigned)(size_t)&As[buf ^ 1][0],
                                 A + (size_t)bm * K + k0n, 128, K);
                tdm_load_tile_2d((unsigned)(size_t)&Ws[buf ^ 1][0],
                                 W + (size_t)bn * K + k0n, 128, K);
                __builtin_amdgcn_s_wait_tensorcnt(2);   // tile `it` landed
            } else {
                __builtin_amdgcn_s_wait_tensorcnt(0);
            }
        }
        __syncthreads();

        v16bf af[2], bfr[4];
#pragma unroll
        for (int i = 0; i < 2; ++i)
            af[i] = load_a_frag((const bf16*)As[buf] + (waveM * 32 + i * 16) * 32, 32, lane);
#pragma unroll
        for (int j = 0; j < 4; ++j)
            bfr[j] = load_b_frag((const bf16*)Ws[buf] + (waveN * 64 + j * 16) * 32, 32, lane);
#pragma unroll
        for (int i = 0; i < 2; ++i)
#pragma unroll
            for (int j = 0; j < 4; ++j)
                acc[i][j] = __builtin_amdgcn_wmma_f32_16x16x32_bf16(
                    false, af[i], false, bfr[j], (short)0, acc[i][j], false, false);
        __syncthreads();   // protect buffer about to be re-filled next iter
    }

    int hi = lane >> 4, n16 = lane & 15;
#pragma unroll
    for (int i = 0; i < 2; ++i) {
#pragma unroll
        for (int j = 0; j < 4; ++j) {
#pragma unroll
            for (int r = 0; r < 8; ++r) {
                int row = bm + waveM * 32 + i * 16 + r + hi * 8;
                int col = bn + waveN * 64 + j * 16 + n16;
                float v = acc[i][j][r];
                if (bias) v += bias[col];
                if (act == 1) v = 0.5f * v * (1.0f + erff(v * 0.70710678118f));
                size_t idx = (size_t)row * N + col;
                if (res)  v += res[idx];
                if (outF) outF[idx] = v;
                if (outB) outB[idx] = (bf16)v;
            }
        }
    }
}

// ---------------------------------------------------------------------------
// Attention: one workgroup = (b, h, 16 query rows). 128 threads = 4 waves.
// ---------------------------------------------------------------------------
__global__ __launch_bounds__(128)
void attention_kernel(const bf16* __restrict__ qkv, bf16* __restrict__ out)
{
    __shared__ __align__(16) bf16 sc[16 * 1024];
    __shared__ __align__(16) bf16 qs[16 * 64];
    __shared__ __align__(16) bf16 vsT[64 * 32];
    __shared__ float red[16 * 8];
    __shared__ float rowmax[16];
    __shared__ float rowsum[16];

    int tid = threadIdx.x, lane = tid & 31, wave = tid >> 5;
    int qb = blockIdx.x, h = blockIdx.y, b = blockIdx.z;
    size_t tok0 = (size_t)b * S_LEN + (size_t)qb * 16;
    const int QKVW = 3 * DMODEL;

    for (int idx = tid; idx < 16 * 64; idx += 128) {
        int r = idx >> 6, c = idx & 63;
        qs[idx] = qkv[(tok0 + r) * QKVW + h * DHEAD + c];
    }
    __syncthreads();

    v8f vzero = {0.f, 0.f, 0.f, 0.f, 0.f, 0.f, 0.f, 0.f};
    int hi = lane >> 4, n16 = lane & 15;

    // ---- scores -----------------------------------------------------------
    for (int t = wave; t < S_LEN / 16; t += 4) {
        v8f acc = vzero;
#pragma unroll
        for (int kc = 0; kc < DHEAD; kc += 32) {
            v16bf a = load_a_frag(qs + kc, 64, lane);
            const bf16* kbase = qkv + ((size_t)b * S_LEN + (size_t)t * 16) * QKVW
                                + DMODEL + h * DHEAD + kc;
            v16bf bb = load_b_frag(kbase, QKVW, lane);
            acc = __builtin_amdgcn_wmma_f32_16x16x32_bf16(
                false, a, false, bb, (short)0, acc, false, false);
        }
#pragma unroll
        for (int r = 0; r < 8; ++r) {
            int m = r + hi * 8;
            sc[m * 1024 + t * 16 + n16] = (bf16)(acc[r] * 0.125f);
        }
    }
    __syncthreads();

    // ---- softmax (unnormalized; rowsum kept) ------------------------------
    {
        int r = tid >> 3, j = tid & 7;
        float mx = -3.0e38f;
        for (int c = j; c < 1024; c += 8) mx = fmaxf(mx, (float)sc[r * 1024 + c]);
        red[r * 8 + j] = mx;
        __syncthreads();
        if (j == 0) {
            float m2 = red[r * 8];
            for (int q = 1; q < 8; ++q) m2 = fmaxf(m2, red[r * 8 + q]);
            rowmax[r] = m2;
        }
        __syncthreads();
        float m2 = rowmax[r], s = 0.f;
        for (int c = j; c < 1024; c += 8) {
            float e = __expf((float)sc[r * 1024 + c] - m2);
            sc[r * 1024 + c] = (bf16)e;
            s += e;
        }
        red[r * 8 + j] = s;
        __syncthreads();
        if (j == 0) {
            float s2 = 0.f;
            for (int q = 0; q < 8; ++q) s2 += red[r * 8 + q];
            rowsum[r] = s2;
        }
        __syncthreads();
    }

    // ---- PV ---------------------------------------------------------------
    v8f oacc = vzero;
    for (int kc = 0; kc < S_LEN; kc += 32) {
        __syncthreads();
        for (int idx = tid; idx < 2048; idx += 128) {
            int k = idx >> 6, c = idx & 63;      // k: key row, c: dh
            vsT[c * 32 + k] = qkv[((size_t)b * S_LEN + kc + k) * QKVW
                                  + 2 * DMODEL + h * DHEAD + c];
        }
        if (kc + 32 < S_LEN) {                   // prefetch next V rows
            const bf16* nv = qkv + ((size_t)b * S_LEN + kc + 32 + (tid >> 2)) * QKVW
                             + 2 * DMODEL + h * DHEAD;
            __builtin_prefetch(nv, 0, 1);
        }
        __syncthreads();
        v16bf a  = load_a_frag(sc + kc, 1024, lane);
        v16bf bb = load_b_frag(vsT + wave * 16 * 32, 32, lane);
        oacc = __builtin_amdgcn_wmma_f32_16x16x32_bf16(
            false, a, false, bb, (short)0, oacc, false, false);
    }
#pragma unroll
    for (int r = 0; r < 8; ++r) {
        int m = r + hi * 8;
        float v = oacc[r] / rowsum[m];
        out[(tok0 + m) * DMODEL + h * DHEAD + wave * 16 + n16] = (bf16)v;
    }
}

// ---------------------------------------------------------------------------
// Embedding + positional
// ---------------------------------------------------------------------------
__global__ __launch_bounds__(256)
void embed_kernel(const int* __restrict__ ids, const float* __restrict__ emb,
                  const float* __restrict__ pos, float* __restrict__ x)
{
    int t = blockIdx.x;
    int id = ids[t];
    int s = t & (S_LEN - 1);
    for (int c = threadIdx.x; c < DMODEL; c += 256)
        x[(size_t)t * DMODEL + c] =
            emb[(size_t)id * DMODEL + c] + pos[(size_t)s * DMODEL + c];
}

// ---------------------------------------------------------------------------
// LayerNorm (block per row of 1024)
// ---------------------------------------------------------------------------
__global__ __launch_bounds__(256)
void layernorm_kernel(const float* __restrict__ x, const float* __restrict__ g,
                      const float* __restrict__ bta,
                      float* __restrict__ outF, bf16* __restrict__ outB)
{
    __shared__ float s1[256], s2[256];
    int row = blockIdx.x, tid = threadIdx.x;
    const float* xr = x + (size_t)row * DMODEL;
    float a = 0.f, q = 0.f;
    for (int c = tid; c < DMODEL; c += 256) { float v = xr[c]; a += v; q += v * v; }
    s1[tid] = a; s2[tid] = q;
    __syncthreads();
    for (int off = 128; off > 0; off >>= 1) {
        if (tid < off) { s1[tid] += s1[tid + off]; s2[tid] += s2[tid + off]; }
        __syncthreads();
    }
    float mean = s1[0] * (1.0f / DMODEL);
    float var  = s2[0] * (1.0f / DMODEL) - mean * mean;
    float rstd = rsqrtf(var + 1e-5f);
    for (int c = tid; c < DMODEL; c += 256) {
        float v = (xr[c] - mean) * rstd * g[c] + bta[c];
        if (outF) outF[(size_t)row * DMODEL + c] = v;
        if (outB) outB[(size_t)row * DMODEL + c] = (bf16)v;
    }
}

// ---------------------------------------------------------------------------
// fp32 -> bf16 conversion
// ---------------------------------------------------------------------------
__global__ __launch_bounds__(256)
void cvt_f32_bf16(const float* __restrict__ in, bf16* __restrict__ out, size_t n)
{
    size_t i = (size_t)blockIdx.x * blockDim.x + threadIdx.x;
    size_t stride = (size_t)gridDim.x * blockDim.x;
    for (; i < n; i += stride) out[i] = (bf16)in[i];
}

// ---------------------------------------------------------------------------
// MXFP8 fake quant: one wave32 per 32-block; pow2 scale; e4m3 RNE; -> bf16.
// ---------------------------------------------------------------------------
__device__ __forceinline__ float e4m3_round(float x)
{
    float a = fabsf(x);
    int e;
    (void)frexpf(a, &e);                  // a = m * 2^e, m in [0.5, 1)
    int se = e - 4;                       // ulp of e4m3 normal at this magnitude
    if (se < -9) se = -9;                 // subnormal floor 2^-9
    float step = ldexpf(1.0f, se);
    float q = rintf(a / step) * step;     // round-to-nearest-even
    q = fminf(q, 448.0f);
    return copysignf(q, x);
}

__global__ __launch_bounds__(256)
void mxfp8_quant_kernel(const float* __restrict__ in, bf16* __restrict__ out,
                        int R, int C, int stride_r, int stride_c)
{
    int lane = threadIdx.x & 31, wave = threadIdx.x >> 5;
    int cb32 = C >> 5;
    long long nblk = (long long)R * cb32;
    long long gb = (long long)blockIdx.x * 8 + wave;
    if (gb >= nblk) return;
    int r = (int)(gb / cb32);
    int c = (int)(gb % cb32) * 32 + lane;
    size_t idx = (size_t)r * stride_r + (size_t)c * stride_c;
    float v = in[idx];
    float a = fabsf(v);
#pragma unroll
    for (int off = 16; off > 0; off >>= 1)
        a = fmaxf(a, __shfl_xor(a, off, 32));
    a = fmaxf(a, 1e-12f);
    float s = exp2f(ceilf(log2f(a * (1.0f / 448.0f))));
    float q = e4m3_round(v / s) * s;
    out[idx] = (bf16)q;
}

// ---------------------------------------------------------------------------
// Host driver
// ---------------------------------------------------------------------------
extern "C" void kernel_launch(void* const* d_in, const int* in_sizes, int n_in,
                              void* d_out, int out_size, void* d_ws, size_t ws_size,
                              hipStream_t stream)
{
    const int*   ids     = (const int*)  d_in[0];
    const float* embed_w = (const float*)d_in[1];
    const float* pos_w   = (const float*)d_in[2];
    const float* ln1_g   = (const float*)d_in[3];
    const float* ln1_b   = (const float*)d_in[4];
    const float* wqkv    = (const float*)d_in[5];
    const float* bqkv    = (const float*)d_in[6];
    const float* wo      = (const float*)d_in[7];
    const float* bo      = (const float*)d_in[8];
    const float* ln2_g   = (const float*)d_in[9];
    const float* ln2_b   = (const float*)d_in[10];
    const float* w1      = (const float*)d_in[11];
    const float* b1      = (const float*)d_in[12];
    const float* w2      = (const float*)d_in[13];
    const float* b2      = (const float*)d_in[14];
    const float* lnf_g   = (const float*)d_in[15];
    const float* lnf_b   = (const float*)d_in[16];
    const float* lm_head = (const float*)d_in[17];

    char* ws = (char*)d_ws;
    size_t off = 0;
    auto alloc = [&](size_t bytes) -> char* {
        char* p = ws + off;
        off += (bytes + 255) & ~(size_t)255;
        return p;
    };

    float* x      = (float*)alloc((size_t)TOKENS * DMODEL * 4);
    float* hbuf   = (float*)alloc((size_t)TOKENS * DMODEL * 4);
    float* ff1    = (float*)alloc((size_t)TOKENS * FFDIM * 4);
    bf16* a_bf    = (bf16*)alloc((size_t)TOKENS * DMODEL * 2);
    bf16* qkv_bf  = (bf16*)alloc((size_t)TOKENS * 3 * DMODEL * 2);
    bf16* attn_bf = (bf16*)alloc((size_t)TOKENS * DMODEL * 2);
    bf16* hq_bf   = (bf16*)alloc((size_t)TOKENS * DMODEL * 2);
    bf16* ff1q_bf = (bf16*)alloc((size_t)TOKENS * FFDIM * 2);
    bf16* wqkv_bf = (bf16*)alloc((size_t)NLAYERS * 3 * DMODEL * DMODEL * 2);
    bf16* wo_bf   = (bf16*)alloc((size_t)NLAYERS * DMODEL * DMODEL * 2);
    bf16* w1q_bf  = (bf16*)alloc((size_t)NLAYERS * FFDIM * DMODEL * 2);
    bf16* w2q_bf  = (bf16*)alloc((size_t)NLAYERS * DMODEL * FFDIM * 2);
    bf16* lmh_bf  = (bf16*)alloc((size_t)VOCAB * DMODEL * 2);

    // --- weight prep (deterministic, every call) ---
    cvt_f32_bf16<<<4096, 256, 0, stream>>>(wqkv, wqkv_bf,
        (size_t)NLAYERS * 3 * DMODEL * DMODEL);
    cvt_f32_bf16<<<2048, 256, 0, stream>>>(wo, wo_bf,
        (size_t)NLAYERS * DMODEL * DMODEL);
    cvt_f32_bf16<<<4096, 256, 0, stream>>>(lm_head, lmh_bf,
        (size_t)VOCAB * DMODEL);
    for (int i = 0; i < NLAYERS; ++i) {
        // w1.T is [D, FF]; blocks along FF (stride D in storage)
        mxfp8_quant_kernel<<<(DMODEL * (FFDIM / 32)) / 8, 256, 0, stream>>>(
            w1 + (size_t)i * FFDIM * DMODEL, w1q_bf + (size_t)i * FFDIM * DMODEL,
            DMODEL, FFDIM, 1, DMODEL);
        // w2.T is [FF, D]; blocks along D (stride FF in storage)
        mxfp8_quant_kernel<<<(FFDIM * (DMODEL / 32)) / 8, 256, 0, stream>>>(
            w2 + (size_t)i * DMODEL * FFDIM, w2q_bf + (size_t)i * DMODEL * FFDIM,
            FFDIM, DMODEL, 1, FFDIM);
    }

    // --- forward ---
    embed_kernel<<<TOKENS, 256, 0, stream>>>(ids, embed_w, pos_w, x);

    for (int i = 0; i < NLAYERS; ++i) {
        layernorm_kernel<<<TOKENS, 256, 0, stream>>>(
            x, ln1_g + i * DMODEL, ln1_b + i * DMODEL, nullptr, a_bf);

        gemm_bf16_kernel<<<dim3(3 * DMODEL / 128, TOKENS / 128), 256, 0, stream>>>(
            a_bf, wqkv_bf + (size_t)i * 3 * DMODEL * DMODEL,
            bqkv + (size_t)i * 3 * DMODEL, nullptr, nullptr, qkv_bf,
            TOKENS, 3 * DMODEL, DMODEL, 0);

        attention_kernel<<<dim3(S_LEN / 16, NHEADS, 2), 128, 0, stream>>>(
            qkv_bf, attn_bf);

        gemm_bf16_kernel<<<dim3(DMODEL / 128, TOKENS / 128), 256, 0, stream>>>(
            attn_bf, wo_bf + (size_t)i * DMODEL * DMODEL,
            bo + i * DMODEL, x, x, nullptr, TOKENS, DMODEL, DMODEL, 0);

        layernorm_kernel<<<TOKENS, 256, 0, stream>>>(
            x, ln2_g + i * DMODEL, ln2_b + i * DMODEL, hbuf, nullptr);

        mxfp8_quant_kernel<<<(TOKENS * (DMODEL / 32)) / 8, 256, 0, stream>>>(
            hbuf, hq_bf, TOKENS, DMODEL, DMODEL, 1);

        gemm_bf16_kernel<<<dim3(FFDIM / 128, TOKENS / 128), 256, 0, stream>>>(
            hq_bf, w1q_bf + (size_t)i * FFDIM * DMODEL,
            b1 + (size_t)i * FFDIM, nullptr, ff1, nullptr,
            TOKENS, FFDIM, DMODEL, 1 /* gelu */);

        mxfp8_quant_kernel<<<(TOKENS * (FFDIM / 32)) / 8, 256, 0, stream>>>(
            ff1, ff1q_bf, TOKENS, FFDIM, FFDIM, 1);

        gemm_bf16_kernel<<<dim3(DMODEL / 128, TOKENS / 128), 256, 0, stream>>>(
            ff1q_bf, w2q_bf + (size_t)i * DMODEL * FFDIM,
            b2 + i * DMODEL, x, x, nullptr, TOKENS, DMODEL, FFDIM, 0);
    }

    layernorm_kernel<<<TOKENS, 256, 0, stream>>>(x, lnf_g, lnf_b, nullptr, a_bf);

    gemm_bf16_kernel<<<dim3(VOCAB / 128, TOKENS / 128), 256, 0, stream>>>(
        a_bf, lmh_bf, nullptr, nullptr, (float*)d_out, nullptr,
        TOKENS, VOCAB, DMODEL, 0);

    (void)in_sizes; (void)n_in; (void)out_size; (void)ws_size;
}
